// FMoE_83279415869765
// MI455X (gfx1250) — matrix operand order repaired and checked by reading, not compiled, not run on hardware
//
#include <hip/hip_runtime.h>

// ---------------------------------------------------------------------------
// FMoE forward for MI455X (gfx1250, wave32, WMMA).
// bf16 tensor-core GEMMs; weights converted once per call to bf16 (134 MB ->
// fits the 192 MB L2), stored transposed [N][K] so both GEMM operands are
// K-contiguous for the WMMA fragment layouts. Tiles are streamed with the
// CDNA5 async global->LDS data mover (ASYNCcnt, GVS addressing: uniform SGPR
// base + loop-invariant VGPR offset) through a 4-buffer LDS ring, prefetch
// depth 3. In-order async completion lets partial counter waits express
// "oldest tile ready"; the drain is peeled so the steady-state loop has no
// branches besides the trip counter.
// ---------------------------------------------------------------------------

typedef __bf16 bf16_t;
typedef __attribute__((ext_vector_type(16))) __bf16 v16bf;
typedef __attribute__((ext_vector_type(8)))  __bf16 v8bf;
typedef __attribute__((ext_vector_type(8)))  float  v8f;

#define N_TOKENS 4096
#define D_MODEL  1024
#define D_HIDDEN 4096
#define N_EXPERT 8

// GEMM tiling: block = 256 threads = 8 wave32, output tile 128x128, K-step 32.
#define BM 128
#define BN 128
#define BK 32
#define LDSROW 40        // 32 halfs + 8 pad (keeps 16B alignment, staggers banks)
#define NBUF 4           // LDS ring buffers
#define DEPTH 3          // tiles in flight
#define BUFBYTES (2 * BM * LDSROW * 2)   // A-plane + B-plane per buffer = 20480B

// ---------------------------------------------------------------------------
// Gate: logits = X @ Wg + bg ; top-2 ; softmax over selected ; scatter to cw.
// One wave32 per token.
// ---------------------------------------------------------------------------
__global__ __launch_bounds__(256) void gate_kernel(
    const float* __restrict__ X, const float* __restrict__ Wg,
    const float* __restrict__ bg, float* __restrict__ cw) {
  const int token = blockIdx.x * (blockDim.x >> 5) + (threadIdx.x >> 5);
  const int lane  = threadIdx.x & 31;
  if (token >= N_TOKENS) return;

  float acc[N_EXPERT];
#pragma unroll
  for (int e = 0; e < N_EXPERT; ++e) acc[e] = 0.0f;

  const float* x = X + (size_t)token * D_MODEL;
  for (int d = lane; d < D_MODEL; d += 32) {
    const float xv = x[d];
    const float* wrow = Wg + (size_t)d * N_EXPERT;
#pragma unroll
    for (int e = 0; e < N_EXPERT; ++e) acc[e] += xv * wrow[e];
  }
#pragma unroll
  for (int e = 0; e < N_EXPERT; ++e) {
#pragma unroll
    for (int off = 16; off > 0; off >>= 1)
      acc[e] += __shfl_xor(acc[e], off, 32);
  }
  if (lane == 0) {
    float v[N_EXPERT];
#pragma unroll
    for (int e = 0; e < N_EXPERT; ++e) v[e] = acc[e] + bg[e];
    int i1 = 0;
#pragma unroll
    for (int e = 1; e < N_EXPERT; ++e) if (v[e] > v[i1]) i1 = e;
    int i2 = (i1 == 0) ? 1 : 0;
#pragma unroll
    for (int e = 0; e < N_EXPERT; ++e)
      if (e != i1 && v[e] > v[i2]) i2 = e;
    // softmax over the two selected logits (max-subtracted)
    const float e2 = __expf(v[i2] - v[i1]);
    const float inv = 1.0f / (1.0f + e2);
    float* row = cw + (size_t)token * N_EXPERT;
#pragma unroll
    for (int e = 0; e < N_EXPERT; ++e) row[e] = 0.0f;
    row[i1] = inv;
    row[i2] = e2 * inv;
  }
}

// ---------------------------------------------------------------------------
// Elementwise fp32 -> bf16 cast.
// ---------------------------------------------------------------------------
__global__ __launch_bounds__(256) void cast_bf16_kernel(
    const float* __restrict__ in, bf16_t* __restrict__ out, int n) {
  int i = blockIdx.x * blockDim.x + threadIdx.x;
  const int stride = gridDim.x * blockDim.x;
  for (; i < n; i += stride) out[i] = (bf16_t)in[i];
}

// ---------------------------------------------------------------------------
// Tiled transpose + cast: in fp32 [R][C] -> out bf16 [C][R], grid.z = expert.
// ---------------------------------------------------------------------------
__global__ __launch_bounds__(256) void transpose_cast_kernel(
    const float* __restrict__ in, bf16_t* __restrict__ out, int R, int C) {
  __shared__ float tile[32][33];
  const float* src = in  + (size_t)blockIdx.z * R * C;
  bf16_t*      dst = out + (size_t)blockIdx.z * R * C;
  const int c0 = blockIdx.x * 32;
  const int r0 = blockIdx.y * 32;
  const int tx = threadIdx.x;   // 0..31
  const int ty = threadIdx.y;   // 0..7
#pragma unroll
  for (int i = 0; i < 32; i += 8)
    tile[ty + i][tx] = src[(size_t)(r0 + ty + i) * C + (c0 + tx)];
  __syncthreads();
#pragma unroll
  for (int i = 0; i < 32; i += 8)
    dst[(size_t)(c0 + ty + i) * R + (r0 + tx)] = (bf16_t)tile[tx][ty + i];
}

__global__ __launch_bounds__(256) void zero_kernel(float* __restrict__ p, int n) {
  int i = blockIdx.x * blockDim.x + threadIdx.x;
  const int stride = gridDim.x * blockDim.x;
  for (; i < n; i += stride) p[i] = 0.0f;
}

// ---------------------------------------------------------------------------
// WMMA GEMM: C[M][N] = A[M][K] * B^T  (B stored [N][K], K-contiguous).
// Block: 256 threads (8 waves, 2x4), tile 128x128, K-step 32.
// 4-deep LDS ring fed by GLOBAL_LOAD_ASYNC_TO_LDS_B128 in GVS mode.
// Epilogue: GELU_EPI ? bf16 gelu(acc+bias) -> Hout
//                    : Yout += cw[row][expert] * (acc+bias)
// ---------------------------------------------------------------------------
template <int KDIM, bool GELU_EPI>
__global__ __launch_bounds__(256, 1) void moe_gemm_kernel(
    const bf16_t* __restrict__ A,
    const bf16_t* __restrict__ B,
    const float*  __restrict__ bias,
    const float*  __restrict__ cw,
    int expert, int ldOut,
    bf16_t* __restrict__ Hout,
    float*  __restrict__ Yout) {
  __shared__ __align__(16) bf16_t sm[NBUF][2][BM * LDSROW];   // 80 KB

  const int tid   = threadIdx.x;
  const int lane  = tid & 31;
  const int wave  = tid >> 5;
  const int waveM = wave >> 2;   // 0..1  (64 rows each)
  const int waveN = wave & 3;    // 0..3  (32 cols each)
  const int tileM = blockIdx.y * BM;
  const int tileN = blockIdx.x * BN;

  v8f acc[4][2];
  const v8f vzero = {0.f, 0.f, 0.f, 0.f, 0.f, 0.f, 0.f, 0.f};
#pragma unroll
  for (int s = 0; s < 4; ++s)
#pragma unroll
    for (int t = 0; t < 2; ++t) acc[s][t] = vzero;

  // Per-thread tile-copy coordinates: 512 chunks of 16B (128 rows x 4), two
  // chunks per thread. Each async b128 op moves 16B/lane directly into LDS,
  // so no staging VGPRs stay live across the WMMA compute.
  const int idx0 = tid, idx1 = tid + 256;
  const int rowC0 = idx0 >> 2, colC0 = (idx0 & 3) * 8;
  const int rowC1 = idx1 >> 2, colC1 = (idx1 & 3) * 8;

  // GVS addressing: loop-invariant per-thread BYTE offsets (32-bit VGPRs);
  // only the uniform SGPR base advances with k0 (scalar add per tile).
  const unsigned offA0 = 2u * (unsigned)((tileM + rowC0) * KDIM + colC0);
  const unsigned offA1 = 2u * (unsigned)((tileM + rowC1) * KDIM + colC1);
  const unsigned offB0 = 2u * (unsigned)((tileN + rowC0) * KDIM + colC0);
  const unsigned offB1 = 2u * (unsigned)((tileN + rowC1) * KDIM + colC1);

  // Flat LDS addresses: low 32 bits of a generic LDS pointer are the LDS
  // byte offset (aperture rule) == what VDST of the async op wants.
  // Ring buffer b is base + b*BUFBYTES (plain VALU add, no cndmask chains).
  const unsigned ldsA0 = (unsigned)(uintptr_t)&sm[0][0][rowC0 * LDSROW + colC0];
  const unsigned ldsA1 = (unsigned)(uintptr_t)&sm[0][0][rowC1 * LDSROW + colC1];
  const unsigned ldsB0 = (unsigned)(uintptr_t)&sm[0][1][rowC0 * LDSROW + colC0];
  const unsigned ldsB1 = (unsigned)(uintptr_t)&sm[0][1][rowC1 * LDSROW + colC1];

  auto issue_tile = [&](int buf, int k0) {
    const unsigned bo = (unsigned)buf * (unsigned)BUFBYTES;
    const unsigned long long baseA =
        (unsigned long long)(uintptr_t)A + 2ull * (unsigned)k0;
    const unsigned long long baseB =
        (unsigned long long)(uintptr_t)B + 2ull * (unsigned)k0;
    asm volatile("global_load_async_to_lds_b128 %0, %1, %2"
                 :: "v"(ldsA0 + bo), "v"(offA0), "s"(baseA) : "memory");
    asm volatile("global_load_async_to_lds_b128 %0, %1, %2"
                 :: "v"(ldsA1 + bo), "v"(offA1), "s"(baseA) : "memory");
    asm volatile("global_load_async_to_lds_b128 %0, %1, %2"
                 :: "v"(ldsB0 + bo), "v"(offB0), "s"(baseB) : "memory");
    asm volatile("global_load_async_to_lds_b128 %0, %1, %2"
                 :: "v"(ldsB1 + bo), "v"(offB1), "s"(baseB) : "memory");
  };

  auto compute_tile = [&](int buf) {
    const bf16_t* sA = sm[buf][0];
    const bf16_t* sB = sm[buf][1];
    const int lr   = lane & 15;
    const int hi16 = lane >> 4;
    // A fragment (16x32 bf16): lanes 0-15 hold K 0-7 & 16-23, lanes 16-31
    // hold K 8-15 & 24-31 (ISA 7.12.2) -> two b128 LDS reads per subtile.
    v16bf afrag[4];
#pragma unroll
    for (int s = 0; s < 4; ++s) {
      const bf16_t* p = sA + (waveM * 64 + s * 16 + lr) * LDSROW + hi16 * 8;
      v8bf lo = *(const v8bf*)(p);
      v8bf hi = *(const v8bf*)(p + 16);
      afrag[s] = __builtin_shufflevector(lo, hi, 0, 1, 2, 3, 4, 5, 6, 7,
                                         8, 9, 10, 11, 12, 13, 14, 15);
    }
    // B fragment (32x16 bf16): lanes = N column, same K order as A
    // (lanes 0-15: K 0-7 & 16-23, lanes 16-31: K 8-15 & 24-31) -> two b128 LDS reads.
    v16bf bfrag[2];
#pragma unroll
    for (int t = 0; t < 2; ++t) {
      const bf16_t* p = sB + (waveN * 32 + t * 16 + lr) * LDSROW + hi16 * 8;
      v8bf lo = *(const v8bf*)(p);
      v8bf hi = *(const v8bf*)(p + 16);
      bfrag[t] = __builtin_shufflevector(lo, hi, 0, 1, 2, 3, 4, 5, 6, 7,
                                         8, 9, 10, 11, 12, 13, 14, 15);
    }
#pragma unroll
    for (int s = 0; s < 4; ++s)
#pragma unroll
      for (int t = 0; t < 2; ++t)
        acc[s][t] = __builtin_amdgcn_wmma_f32_16x16x32_bf16(
            false, afrag[s], false, bfrag[t], (short)0, acc[s][t],
            false, false);
  };

  constexpr int NK = KDIM / BK;   // 32 or 128
  static_assert(NK >= DEPTH + 1, "K too small for pipeline depth");

  // Prologue: put DEPTH tiles in flight.
#pragma unroll
  for (int p = 0; p < DEPTH; ++p) issue_tile(p, p * BK);

  // Steady state: branch-free body (wait for oldest of 3 in-flight tiles).
#pragma unroll 1
  for (int kt = 0; kt < NK - DEPTH; ++kt) {
    asm volatile("s_wait_asynccnt 0x8" ::: "memory");
    __syncthreads();            // publish tile kt; everyone done reading kt-1
    issue_tile((kt + DEPTH) & (NBUF - 1), (kt + DEPTH) * BK);
    compute_tile(kt & (NBUF - 1));
  }

  // Peeled drain: last DEPTH tiles, constant waits / buffer indices.
  {
    constexpr int k0 = NK - DEPTH;
    asm volatile("s_wait_asynccnt 0x8" ::: "memory");
    __syncthreads();
    compute_tile(k0 & (NBUF - 1));
    asm volatile("s_wait_asynccnt 0x4" ::: "memory");
    __syncthreads();
    compute_tile((k0 + 1) & (NBUF - 1));
    asm volatile("s_wait_asynccnt 0x0" ::: "memory");
    __syncthreads();
    compute_tile((k0 + 2) & (NBUF - 1));
  }

  // Epilogue. C/D layout: VGPR r -> row r (lanes 0-15) / row r+8 (lanes 16-31),
  // N = lane&15.
  const int lr   = lane & 15;
  const int hi16 = lane >> 4;
#pragma unroll
  for (int s = 0; s < 4; ++s) {
    const int rowBase = tileM + waveM * 64 + s * 16 + hi16 * 8;
#pragma unroll
    for (int t = 0; t < 2; ++t) {
      const int col = tileN + waveN * 32 + t * 16 + lr;
      const float bv = bias[col];
#pragma unroll
      for (int r = 0; r < 8; ++r) {
        const int row = rowBase + r;
        const float val = acc[s][t][r] + bv;
        if constexpr (GELU_EPI) {
          // tanh-approx GELU, branch-free: 0.5*(1+tanh(u)) == sigmoid(2u)
          const float u = val * (0.7978845608f + 0.0356774081f * val * val);
          const float g =
              val * __builtin_amdgcn_rcpf(1.0f + __expf(-2.0f * u));
          Hout[(size_t)row * ldOut + col] = (bf16_t)g;
        } else {
          const float w = cw[(size_t)row * N_EXPERT + expert];
          float* o = Yout + (size_t)row * ldOut + col;
          *o += w * val;
        }
      }
    }
  }
}

// ---------------------------------------------------------------------------
// Host-side orchestration (all on `stream`, graph-capture safe).
// Workspace: Xbf 8MB | W1bf 64MB | W2bf 64MB | Hbf 32MB | cw 128KB  ~168MB.
// ---------------------------------------------------------------------------
extern "C" void kernel_launch(void* const* d_in, const int* in_sizes, int n_in,
                              void* d_out, int out_size, void* d_ws, size_t ws_size,
                              hipStream_t stream) {
  const float* X  = (const float*)d_in[0];
  // d_in[1] = attn_weights: unused by the reference computation.
  const float* Wg = (const float*)d_in[2];
  const float* bg = (const float*)d_in[3];
  const float* W1 = (const float*)d_in[4];
  const float* b1 = (const float*)d_in[5];
  const float* W2 = (const float*)d_in[6];
  const float* b2 = (const float*)d_in[7];
  float* out = (float*)d_out;

  char* ws = (char*)d_ws;
  bf16_t* Xbf  = (bf16_t*)ws;  ws += (size_t)N_TOKENS * D_MODEL * sizeof(bf16_t);
  bf16_t* W1bf = (bf16_t*)ws;  ws += (size_t)N_EXPERT * D_MODEL * D_HIDDEN * sizeof(bf16_t);
  bf16_t* W2bf = (bf16_t*)ws;  ws += (size_t)N_EXPERT * D_MODEL * D_HIDDEN * sizeof(bf16_t);
  bf16_t* Hbf  = (bf16_t*)ws;  ws += (size_t)N_TOKENS * D_HIDDEN * sizeof(bf16_t);
  float*  cw   = (float*)ws;   ws += (size_t)N_TOKENS * N_EXPERT * sizeof(float);

  // 1) Gate -> combine weights cw[N][E]
  gate_kernel<<<N_TOKENS / 8, 256, 0, stream>>>(X, Wg, bg, cw);

  // 2) Precision staging: X -> bf16; W1/W2 -> bf16 transposed [N][K]
  cast_bf16_kernel<<<2048, 256, 0, stream>>>(X, Xbf, N_TOKENS * D_MODEL);
  transpose_cast_kernel<<<dim3(D_HIDDEN / 32, D_MODEL / 32, N_EXPERT),
                          dim3(32, 8), 0, stream>>>(W1, W1bf, D_MODEL, D_HIDDEN);
  transpose_cast_kernel<<<dim3(D_MODEL / 32, D_HIDDEN / 32, N_EXPERT),
                          dim3(32, 8), 0, stream>>>(W2, W2bf, D_HIDDEN, D_MODEL);

  // 3) out = 0 (harness poisons d_out)
  zero_kernel<<<2048, 256, 0, stream>>>(out, N_TOKENS * D_MODEL);

  // 4) Per-expert FFN: h = gelu(X W1 + b1); out += cw[:,e] * (h W2 + b2)
  for (int e = 0; e < N_EXPERT; ++e) {
    moe_gemm_kernel<D_MODEL, true>
        <<<dim3(D_HIDDEN / BN, N_TOKENS / BM), 256, 0, stream>>>(
            Xbf, W1bf + (size_t)e * D_MODEL * D_HIDDEN,
            b1 + (size_t)e * D_HIDDEN, nullptr, e, D_HIDDEN, Hbf, nullptr);
    moe_gemm_kernel<D_HIDDEN, false>
        <<<dim3(D_MODEL / BN, N_TOKENS / BM), 256, 0, stream>>>(
            Hbf, W2bf + (size_t)e * D_MODEL * D_HIDDEN,
            b2 + (size_t)e * D_MODEL, cw, e, D_MODEL, nullptr, out);
  }
}
